// DET_78683800863456
// MI455X (gfx1250) — matrix operand order repaired and checked by reading, not compiled, not run on hardware
//
#include <hip/hip_runtime.h>
#include <math.h>

// ---------------- model dims ----------------
#define BS     8
#define NCH    64
#define NSEC   16
#define FEAT   200
#define L_TOK  1024         // NCH*NSEC
#define D      512
#define DEPTH  8
#define HEADS  8
#define DH     64
#define MLP    2048
#define NSEQ   1025         // tokens + cls
#define NPAD   1056         // padded to multiple of 32 (33 key blocks, 66 q tiles)
#define ROWS   (BS*NPAD)    // 8448 rows in the padded activation matrix
#define LN_EPS 1e-5f

// ---------------- vector / wmma types ----------------
typedef __bf16 bf16_t;
typedef __attribute__((ext_vector_type(16))) __bf16 bf16x16;
typedef __attribute__((ext_vector_type(8)))  __bf16 bf16x8;
typedef __attribute__((ext_vector_type(8)))  float  f32x8;
typedef __attribute__((ext_vector_type(4)))  unsigned int u32x4;
typedef __attribute__((ext_vector_type(4)))  int i32x4;
typedef __attribute__((ext_vector_type(8)))  int i32x8;

#define CAT16(lo, hi) __builtin_shufflevector((lo), (hi), 0,1,2,3,4,5,6,7,8,9,10,11,12,13,14,15)

#if __has_builtin(__builtin_amdgcn_tensor_load_to_lds)
#define HAVE_TDM 1
#else
#define HAVE_TDM 0
#endif

__device__ __forceinline__ bf16x8 ld8(const bf16_t* p) { return *(const bf16x8*)p; }

__device__ __forceinline__ f32x8 wmma_bf16(bf16x16 a, bf16x16 b, f32x8 c) {
    // D = A(16x32) * B(32x16) + C, bf16 in, f32 accum
    return __builtin_amdgcn_wmma_f32_16x16x32_bf16(
        /*neg_a=*/false, a, /*neg_b=*/false, b,
        /*c_mod=*/(short)0, c, /*reuse_a=*/false, /*reuse_b=*/false);
}

__device__ __forceinline__ float wave_sum(float s) {
#pragma unroll
    for (int m = 1; m < 32; m <<= 1) s += __shfl_xor(s, m, 32);
    return s;
}

__device__ __forceinline__ void wait_tensorcnt0() {
#if __has_builtin(__builtin_amdgcn_s_wait_tensorcnt)
    __builtin_amdgcn_s_wait_tensorcnt(0);
#else
    asm volatile("s_wait_tensorcnt 0" ::: "memory");
#endif
}

// Issue a TDM 2D tile load: 64 rows x 32 bf16 (row stride = K elems) -> LDS (compact).
// Descriptor packing per CDNA5 ISA 8.3/8.4 (count=1, type=2, data_size=2B).
__device__ __forceinline__ void tdm_load_tile(const bf16_t* gptr, unsigned lds_off, int K) {
#if HAVE_TDM
    unsigned long long ga = (unsigned long long)(uintptr_t)gptr;
    u32x4 g0;
    g0[0] = 1u;                                               // count=1, user mode
    g0[1] = lds_off;                                          // lds_addr (bytes)
    g0[2] = (unsigned)(ga & 0xffffffffull);                   // global_addr[31:0]
    g0[3] = (unsigned)((ga >> 32) & 0x01ffffffull) | 0x80000000u;  // addr[56:32] | type=2
    i32x8 g1;
    g1[0] = 0x00010000;                                       // data_size=1 (2 bytes)
    g1[1] = (int)(((unsigned)K & 0xffffu) << 16);             // tensor_dim0 lo16
    g1[2] = (int)((((unsigned)K >> 16) & 0xffffu) | (64u << 16)); // td0 hi | tensor_dim1=64
    g1[3] = (int)(32u << 16);                                 // td1 hi=0 | tile_dim0=32
    g1[4] = 64;                                               // tile_dim1=64, tile_dim2=0
    g1[5] = K;                                                // tensor_dim0_stride lo32
    g1[6] = 0;                                                // stride0 hi | stride1 lo
    g1[7] = 0;                                                // stride1 hi
    i32x4 z4 = {};
#if __clang_major__ >= 23
    i32x8 z8 = {};
    __builtin_amdgcn_tensor_load_to_lds(g0, g1, z4, z4, z8, 0);
#else
    __builtin_amdgcn_tensor_load_to_lds(g0, g1, z4, z4, 0);
#endif
#else
    (void)gptr; (void)lds_off; (void)K;
#endif
}

// ---------------- weight cast + transpose: src[K,N] f32 -> dst[N,K] bf16 ----------------
__global__ void tcast_kernel(const float* __restrict__ src, bf16_t* __restrict__ dst,
                             int K, int N) {
    int idx = blockIdx.x * 256 + threadIdx.x;
    if (idx >= K * N) return;
    int k = idx / N, n = idx - k * N;
    dst[(size_t)n * K + k] = (bf16_t)src[idx];
}

// ---------------- embedding: circular conv1d(k=3) + temporal embed ----------------
__global__ __launch_bounds__(256)
void embed_kernel(const float* __restrict__ x, const float* __restrict__ conv_w,
                  const float* __restrict__ tem, float* __restrict__ h) {
    __shared__ float feats[3 * FEAT];
    int bl = blockIdx.x;           // b*1024 + l
    int b = bl >> 10, l = bl & 1023;
    int tid = threadIdx.x;
    for (int i = tid; i < 3 * FEAT; i += 256) {
        int t = i / FEAT, f = i - t * FEAT;
        int ls = (l + t - 1 + L_TOK) & (L_TOK - 1);   // circular pad
        feats[t * FEAT + f] = x[((size_t)b * L_TOK + ls) * FEAT + f];
    }
    __syncthreads();
    for (int d = tid; d < D; d += 256) {
        const float* w = conv_w + (size_t)d * (FEAT * 3);   // [d][f][t]
        float acc = 0.f;
#pragma unroll 4
        for (int f = 0; f < FEAT; ++f) {
            acc += w[f * 3 + 0] * feats[f]
                 + w[f * 3 + 1] * feats[FEAT + f]
                 + w[f * 3 + 2] * feats[2 * FEAT + f];
        }
        h[((size_t)b * NPAD + 1 + l) * D + d] = acc + tem[(l & 15) * D + d];
    }
}

// cls token into row 0, zeros into pad rows 1025..1055
__global__ void init_rows_kernel(const float* __restrict__ cls, float* __restrict__ h) {
    int b = blockIdx.x >> 5, r = blockIdx.x & 31;
    int tid = threadIdx.x;
    if (r == 0) {
        size_t base = (size_t)b * NPAD * D;
        for (int c = tid; c < D; c += 256) h[base + c] = cls[c];
    } else {
        size_t base = ((size_t)b * NPAD + 1024 + r) * D;   // rows 1025..1055
        for (int c = tid; c < D; c += 256) h[base + c] = 0.f;
    }
}

// ---------------- LayerNorm + cast to bf16 (one wave per row) ----------------
__global__ __launch_bounds__(256)
void ln_cast_kernel(const float* __restrict__ h, const float* __restrict__ g,
                    const float* __restrict__ b, bf16_t* __restrict__ out) {
    int row = blockIdx.x * 8 + (threadIdx.x >> 5);
    int lane = threadIdx.x & 31;
    int n = row % NPAD;
    const float* hp = h + (size_t)row * D;
    bf16_t* op = out + (size_t)row * D;
    if (n >= NSEQ) {          // pad row -> zeros (keeps all WMMA inputs finite)
        for (int i = lane; i < D; i += 32) op[i] = (bf16_t)0.0f;
        return;
    }
    float v[16]; float s = 0.f;
#pragma unroll
    for (int i = 0; i < 16; ++i) { v[i] = hp[lane + i * 32]; s += v[i]; }
    float mean = wave_sum(s) * (1.f / D);
    float var = 0.f;
#pragma unroll
    for (int i = 0; i < 16; ++i) { float d0 = v[i] - mean; var += d0 * d0; }
    var = wave_sum(var) * (1.f / D);
    float inv = rsqrtf(var + LN_EPS);
#pragma unroll
    for (int i = 0; i < 16; ++i) {
        int c = lane + i * 32;
        op[c] = (bf16_t)((v[i] - mean) * inv * g[c] + b[c]);
    }
}

// ---------------- generic WMMA GEMM: out[M,N] = A[M,K](bf16) * BT[N,K](bf16) ----------------
// B tile (64 cols x 32 k, 4KB) staged into LDS by the Tensor Data Mover, double-buffered:
// wave 0 issues the TDM DMA for k-step t+1 while all 8 waves run WMMAs on k-step t.
// MODE 0: outF = acc                MODE 1: outF += acc + bias       MODE 2: outB = bf16(gelu(acc+b))
template<int MODE>
__global__ __launch_bounds__(256)
void gemm_bf16_kernel(const bf16_t* __restrict__ A, const bf16_t* __restrict__ BT,
                      const float* __restrict__ bias, float* __restrict__ outF,
                      bf16_t* __restrict__ outB, int K, int ldo) {
#if HAVE_TDM
    __shared__ __align__(16) bf16_t Bs[2][64 * 32];
#endif
    int wave = threadIdx.x >> 5;
    int lane = threadIdx.x & 31;
    int lo = lane & 15, hi = lane >> 4;
    int row0 = blockIdx.x * 128 + wave * 16;
    int col0 = blockIdx.y * 64;

    f32x8 acc[4] = {};
    const bf16_t* Ap = A + (size_t)(row0 + lo) * K;

#if HAVE_TDM
    const bf16_t* Bg = BT + (size_t)col0 * K;
    if (wave == 0) {
        tdm_load_tile(Bg, (unsigned)(uintptr_t)(&Bs[0][0]), K);
        wait_tensorcnt0();
    }
    __syncthreads();
    int buf = 0;
    for (int k0 = 0; k0 < K; k0 += 32) {
        if (wave == 0 && (k0 + 32) < K)
            tdm_load_tile(Bg + k0 + 32, (unsigned)(uintptr_t)(&Bs[buf ^ 1][0]), K);
        __builtin_prefetch(Ap + k0 + 64, 0, 1);   // next A chunk -> global_prefetch
        bf16x16 af = CAT16(ld8(Ap + k0 + hi * 8), ld8(Ap + k0 + 16 + hi * 8));
        const bf16_t* Bb = &Bs[buf][0];
#pragma unroll
        for (int nt = 0; nt < 4; ++nt) {
            const bf16_t* Bp = Bb + (nt * 16 + lo) * 32 + hi * 8;
            acc[nt] = wmma_bf16(af, CAT16(ld8(Bp), ld8(Bp + 16)), acc[nt]);
        }
        if (wave == 0) wait_tensorcnt0();
        __syncthreads();
        buf ^= 1;
    }
#else
    for (int k0 = 0; k0 < K; k0 += 32) {
        bf16x16 af = CAT16(ld8(Ap + k0 + hi * 8), ld8(Ap + k0 + 16 + hi * 8));
#pragma unroll
        for (int nt = 0; nt < 4; ++nt) {
            const bf16_t* Bp = BT + (size_t)(col0 + nt * 16 + lo) * K + k0 + hi * 8;
            acc[nt] = wmma_bf16(af, CAT16(ld8(Bp), ld8(Bp + 16)), acc[nt]);
        }
    }
#endif

#pragma unroll
    for (int nt = 0; nt < 4; ++nt) {
        int col = col0 + nt * 16 + lo;
        float bv = (MODE == 0) ? 0.f : bias[col];
#pragma unroll
        for (int j = 0; j < 8; ++j) {
            int row = row0 + j + hi * 8;
            float v = acc[nt][j] + bv;
            if (MODE == 0) {
                outF[(size_t)row * ldo + col] = v;
            } else if (MODE == 1) {
                outF[(size_t)row * ldo + col] += v;          // residual add
            } else {
                float ge = 0.5f * v * (1.f + erff(v * 0.70710678118654752f));
                outB[(size_t)row * ldo + col] = (bf16_t)ge;
            }
        }
    }
}

// ---------------- RoPE + pack q/k (bf16) and vT (bf16, [bh][64][NPAD]) ----------------
__global__ __launch_bounds__(256)
void rope_pack_kernel(const float* __restrict__ qkv, bf16_t* __restrict__ qb,
                      bf16_t* __restrict__ kb, bf16_t* __restrict__ vT) {
    int idx = blockIdx.x * 256 + threadIdx.x;   // B*H*NPAD*32 threads
    int i = idx & 31; int t = idx >> 5;
    int n = t % NPAD; t /= NPAD;
    int hh = t & 7; int b = t >> 3;
    size_t bh = (size_t)(b * HEADS + hh);
    bf16_t* qo = qb + (bh * NPAD + n) * DH + 2 * i;
    bf16_t* ko = kb + (bh * NPAD + n) * DH + 2 * i;
    size_t vb = (bh * DH + 2 * i) * NPAD + n;
    if (n >= NSEQ) {
        qo[0] = qo[1] = ko[0] = ko[1] = (bf16_t)0.0f;
        vT[vb] = (bf16_t)0.0f; vT[vb + NPAD] = (bf16_t)0.0f;
        return;
    }
    const float* src = qkv + ((size_t)b * NPAD + n) * (3 * D) + hh * DH + 2 * i;
    float q0 = src[0], q1 = src[1];
    float k0 = src[D], k1 = src[D + 1];
    float v0 = src[2 * D], v1 = src[2 * D + 1];
    float inv = __expf(-0.28782313662425572f * (float)i);   // 10000^(-2i/64)
    float ang = (float)n * inv;
    float s, c; __sincosf(ang, &s, &c);
    qo[0] = (bf16_t)(q0 * c - q1 * s);  qo[1] = (bf16_t)(q1 * c + q0 * s);
    ko[0] = (bf16_t)(k0 * c - k1 * s);  ko[1] = (bf16_t)(k1 * c + k0 * s);
    vT[vb] = (bf16_t)v0;  vT[vb + NPAD] = (bf16_t)v1;
}

// ---------------- flash-style attention, one wave per (b, h, 16-query tile) ----------------
__global__ __launch_bounds__(32)
void attn_kernel(const bf16_t* __restrict__ qb, const bf16_t* __restrict__ kb,
                 const bf16_t* __restrict__ vT, bf16_t* __restrict__ aout) {
    __shared__ __align__(16) bf16_t pl[16 * 32];   // P tile staged C-layout -> A-layout
    int bid = blockIdx.x;
    int qt = bid % (NPAD / 16); bid /= (NPAD / 16);
    int hh = bid & 7, b = bid >> 3;
    int lane = threadIdx.x;
    int lo = lane & 15, hi = lane >> 4;
    size_t bh = (size_t)(b * HEADS + hh);

    // Q fragments (rows = queries, K = dh)
    const bf16_t* qp = qb + (bh * NPAD + qt * 16 + lo) * DH;
    bf16x16 aq0 = CAT16(ld8(qp +  0 + hi * 8), ld8(qp + 16 + hi * 8));  // dh 0..31
    bf16x16 aq1 = CAT16(ld8(qp + 32 + hi * 8), ld8(qp + 48 + hi * 8));  // dh 32..63

    f32x8 o[4] = {};
    float mrow[8], ssum[8];
#pragma unroll
    for (int j = 0; j < 8; ++j) { mrow[j] = -3e38f; ssum[j] = 0.f; }

    const bf16_t* kbase = kb + bh * NPAD * DH;
    const bf16_t* vbase = vT + bh * DH * NPAD;

    for (int key0 = 0; key0 < NPAD; key0 += 32) {
        // ---- scores: two 16x16 tiles, each over dh in two k-steps ----
        f32x8 s0 = {}, s1 = {};
        {
            const bf16_t* kp = kbase + (size_t)(key0 + lo) * DH + hi * 8;
            s0 = wmma_bf16(aq0, CAT16(ld8(kp),      ld8(kp + 16)),      s0);
            s0 = wmma_bf16(aq1, CAT16(ld8(kp + 32), ld8(kp + 32 + 16)), s0);
        }
        {
            const bf16_t* kp = kbase + (size_t)(key0 + 16 + lo) * DH + hi * 8;
            s1 = wmma_bf16(aq0, CAT16(ld8(kp),      ld8(kp + 16)),      s1);
            s1 = wmma_bf16(aq1, CAT16(ld8(kp + 32), ld8(kp + 32 + 16)), s1);
        }
        // ---- scale + key mask (finite sentinel) ----
        bool ok0 = (key0 + lo) < NSEQ;
        bool ok1 = (key0 + 16 + lo) < NSEQ;
        float mx[8];
#pragma unroll
        for (int j = 0; j < 8; ++j) {
            float a_ = ok0 ? s0[j] * 0.125f : -3e38f;
            float b_ = ok1 ? s1[j] * 0.125f : -3e38f;
            s0[j] = a_; s1[j] = b_;
            mx[j] = fmaxf(a_, b_);
        }
        // row max across the 16 lanes of each half
#pragma unroll
        for (int msk = 1; msk <= 8; msk <<= 1)
#pragma unroll
            for (int j = 0; j < 8; ++j) mx[j] = fmaxf(mx[j], __shfl_xor(mx[j], msk, 32));
        // ---- online softmax update ----
        float f[8], rs[8];
        f32x8 p0, p1;
#pragma unroll
        for (int j = 0; j < 8; ++j) {
            float mn = fmaxf(mrow[j], mx[j]);
            f[j] = __expf(mrow[j] - mn);
            mrow[j] = mn;
            p0[j] = __expf(s0[j] - mn);
            p1[j] = __expf(s1[j] - mn);
            rs[j] = p0[j] + p1[j];
        }
#pragma unroll
        for (int msk = 1; msk <= 8; msk <<= 1)
#pragma unroll
            for (int j = 0; j < 8; ++j) rs[j] += __shfl_xor(rs[j], msk, 32);
#pragma unroll
        for (int j = 0; j < 8; ++j) {
            ssum[j] = ssum[j] * f[j] + rs[j];
            o[0][j] *= f[j]; o[1][j] *= f[j]; o[2][j] *= f[j]; o[3][j] *= f[j];
        }
        // ---- stage P through LDS: C-layout store, A-layout load ----
#pragma unroll
        for (int j = 0; j < 8; ++j) {
            pl[(j + hi * 8) * 32 + lo]      = (bf16_t)p0[j];
            pl[(j + hi * 8) * 32 + 16 + lo] = (bf16_t)p1[j];
        }
        __syncthreads();
        asm volatile("s_wait_dscnt 0" ::: "memory");
        const bf16_t* pr = pl + lo * 32;
        bf16x16 ap = CAT16(ld8(pr + hi * 8), ld8(pr + 16 + hi * 8));
        __syncthreads();
        // ---- P @ V : 4 dh tiles ----
#pragma unroll
        for (int t4 = 0; t4 < 4; ++t4) {
            const bf16_t* vr = vbase + (size_t)(t4 * 16 + lo) * NPAD + key0 + hi * 8;
            o[t4] = wmma_bf16(ap, CAT16(ld8(vr), ld8(vr + 16)), o[t4]);
        }
    }
    // ---- normalize and store bf16 into [row, 512] (col = h*64 + d) ----
    float inv[8];
#pragma unroll
    for (int j = 0; j < 8; ++j) inv[j] = 1.0f / ssum[j];
    size_t obase = ((size_t)b * NPAD + qt * 16) * D + hh * DH;
#pragma unroll
    for (int t4 = 0; t4 < 4; ++t4)
#pragma unroll
        for (int j = 0; j < 8; ++j)
            aout[obase + (size_t)(j + hi * 8) * D + t4 * 16 + lo] = (bf16_t)(o[t4][j] * inv[j]);
}

// ---------------- final double LayerNorm on cls rows ----------------
__global__ __launch_bounds__(32)
void final_kernel(const float* __restrict__ h,
                  const float* __restrict__ fg, const float* __restrict__ fb,
                  const float* __restrict__ og, const float* __restrict__ ob,
                  float* __restrict__ out) {
    int b = blockIdx.x, lane = threadIdx.x;
    const float* hp = h + (size_t)b * NPAD * D;   // row 0 = cls
    float v[16]; float s = 0.f;
#pragma unroll
    for (int i = 0; i < 16; ++i) { v[i] = hp[lane + i * 32]; s += v[i]; }
    float mean = wave_sum(s) * (1.f / D);
    float var = 0.f;
#pragma unroll
    for (int i = 0; i < 16; ++i) { float d0 = v[i] - mean; var += d0 * d0; }
    float inv = rsqrtf(wave_sum(var) * (1.f / D) + LN_EPS);
    s = 0.f;
#pragma unroll
    for (int i = 0; i < 16; ++i) {
        int c = lane + i * 32;
        v[i] = (v[i] - mean) * inv * fg[c] + fb[c];
        s += v[i];
    }
    float mean2 = wave_sum(s) * (1.f / D);
    var = 0.f;
#pragma unroll
    for (int i = 0; i < 16; ++i) { float d0 = v[i] - mean2; var += d0 * d0; }
    float inv2 = rsqrtf(wave_sum(var) * (1.f / D) + LN_EPS);
#pragma unroll
    for (int i = 0; i < 16; ++i) {
        int c = lane + i * 32;
        out[(size_t)b * D + c] = (v[i] - mean2) * inv2 * og[c] + ob[c];
    }
}

// ---------------- host orchestration ----------------
extern "C" void kernel_launch(void* const* d_in, const int* in_sizes, int n_in,
                              void* d_out, int out_size, void* d_ws, size_t ws_size,
                              hipStream_t stream) {
    const float* x      = (const float*)d_in[0];
    const float* conv_w = (const float*)d_in[1];
    const float* cls    = (const float*)d_in[2];
    const float* tem    = (const float*)d_in[3];
    const float* ln1_g  = (const float*)d_in[4];
    const float* ln1_b  = (const float*)d_in[5];
    const float* wqkv   = (const float*)d_in[6];
    const float* wo     = (const float*)d_in[7];
    const float* bo     = (const float*)d_in[8];
    const float* ln2_g  = (const float*)d_in[9];
    const float* ln2_b  = (const float*)d_in[10];
    const float* w1     = (const float*)d_in[11];
    const float* b1     = (const float*)d_in[12];
    const float* w2     = (const float*)d_in[13];
    const float* b2     = (const float*)d_in[14];
    const float* lnf_g  = (const float*)d_in[15];
    const float* lnf_b  = (const float*)d_in[16];
    const float* lno_g  = (const float*)d_in[17];
    const float* lno_b  = (const float*)d_in[18];
    float* out = (float*)d_out;

    size_t off = 0;
    auto alloc = [&](size_t bytes) -> void* {
        void* p = (char*)d_ws + off;
        off += (bytes + 255) & ~(size_t)255;
        return p;
    };
    float*  h     = (float*) alloc((size_t)ROWS * D * 4);
    bf16_t* ybf   = (bf16_t*)alloc((size_t)ROWS * D * 2);
    float*  qkv   = (float*) alloc((size_t)ROWS * 3 * D * 4);
    bf16_t* qb    = (bf16_t*)alloc((size_t)BS * HEADS * NPAD * DH * 2);
    bf16_t* kbuf  = (bf16_t*)alloc((size_t)BS * HEADS * NPAD * DH * 2);
    bf16_t* vT    = (bf16_t*)alloc((size_t)BS * HEADS * DH * NPAD * 2);
    bf16_t* aoutb = (bf16_t*)alloc((size_t)ROWS * D * 2);
    bf16_t* mid   = (bf16_t*)alloc((size_t)ROWS * MLP * 2);
    bf16_t* wqkvT = (bf16_t*)alloc((size_t)DEPTH * 3 * D * D * 2);
    bf16_t* woT   = (bf16_t*)alloc((size_t)DEPTH * D * D * 2);
    bf16_t* w1T   = (bf16_t*)alloc((size_t)DEPTH * MLP * D * 2);
    bf16_t* w2T   = (bf16_t*)alloc((size_t)DEPTH * D * MLP * 2);
    (void)ws_size; (void)n_in; (void)in_sizes; (void)out_size;

    // weight casts (W -> W^T bf16)
    for (int i = 0; i < DEPTH; ++i) {
        tcast_kernel<<<(D * 3 * D + 255) / 256, 256, 0, stream>>>(
            wqkv + (size_t)i * D * 3 * D, wqkvT + (size_t)i * 3 * D * D, D, 3 * D);
        tcast_kernel<<<(D * D + 255) / 256, 256, 0, stream>>>(
            wo + (size_t)i * D * D, woT + (size_t)i * D * D, D, D);
        tcast_kernel<<<(D * MLP + 255) / 256, 256, 0, stream>>>(
            w1 + (size_t)i * D * MLP, w1T + (size_t)i * MLP * D, D, MLP);
        tcast_kernel<<<(MLP * D + 255) / 256, 256, 0, stream>>>(
            w2 + (size_t)i * MLP * D, w2T + (size_t)i * D * MLP, MLP, D);
    }

    // embedding
    embed_kernel<<<BS * L_TOK, 256, 0, stream>>>(x, conv_w, tem, h);
    init_rows_kernel<<<BS * 32, 256, 0, stream>>>(cls, h);

    for (int i = 0; i < DEPTH; ++i) {
        ln_cast_kernel<<<ROWS / 8, 256, 0, stream>>>(h, ln1_g + i * D, ln1_b + i * D, ybf);
        gemm_bf16_kernel<0><<<dim3(ROWS / 128, (3 * D) / 64), 256, 0, stream>>>(
            ybf, wqkvT + (size_t)i * 3 * D * D, nullptr, qkv, nullptr, D, 3 * D);
        rope_pack_kernel<<<(BS * HEADS * NPAD * 32) / 256, 256, 0, stream>>>(qkv, qb, kbuf, vT);
        attn_kernel<<<BS * HEADS * (NPAD / 16), 32, 0, stream>>>(qb, kbuf, vT, aoutb);
        gemm_bf16_kernel<1><<<dim3(ROWS / 128, D / 64), 256, 0, stream>>>(
            aoutb, woT + (size_t)i * D * D, bo + i * D, h, nullptr, D, D);
        ln_cast_kernel<<<ROWS / 8, 256, 0, stream>>>(h, ln2_g + i * D, ln2_b + i * D, ybf);
        gemm_bf16_kernel<2><<<dim3(ROWS / 128, MLP / 64), 256, 0, stream>>>(
            ybf, w1T + (size_t)i * MLP * D, b1 + i * MLP, nullptr, mid, D, MLP);
        gemm_bf16_kernel<1><<<dim3(ROWS / 128, D / 64), 256, 0, stream>>>(
            mid, w2T + (size_t)i * D * MLP, b2 + i * D, h, nullptr, MLP, D);
    }

    final_kernel<<<BS, 32, 0, stream>>>(h, lnf_g, lnf_b, lno_g, lno_b, out);
}
